// RNAGAT_43026982371536
// MI455X (gfx1250) — matrix operand order repaired and checked by reading, not compiled, not run on hardware
//
#include <hip/hip_runtime.h>

#define NN 50000
#define NE 800000
#define D 128
#define HE 32
#define DEPTH 8

typedef __attribute__((ext_vector_type(16))) __bf16    v16bf;
typedef __attribute__((ext_vector_type(8)))  float     v8f;
typedef __attribute__((ext_vector_type(4)))  unsigned  u32x4;

union Frag { v16bf v; u32x4 q[2]; };

__device__ __forceinline__ unsigned short f2bf(float x) {
  unsigned u = __float_as_uint(x);
  u += 0x7FFFu + ((u >> 16) & 1u);          // round-to-nearest-even
  return (unsigned short)(u >> 16);
}

// order-preserving float<->uint encoding for atomicMax
__device__ __forceinline__ unsigned enc_f32(float x) {
  unsigned b = __float_as_uint(x);
  return (b & 0x80000000u) ? ~b : (b | 0x80000000u);
}
__device__ __forceinline__ float dec_f32(unsigned u) {
  unsigned b = (u & 0x80000000u) ? (u & 0x7FFFFFFFu) : ~u;
  return __uint_as_float(b);
}

// ---------------- node embedding: h_bf16[n][c] = concat(Wseq[seq], Wsec[sec]) ----
__global__ __launch_bounds__(256)
void embed_kernel(const int* __restrict__ seq, const int* __restrict__ sec,
                  const float* __restrict__ Wseq, const float* __restrict__ Wsec,
                  unsigned short* __restrict__ hbf) {
  int i = blockIdx.x * 256 + threadIdx.x;     // over NN*D (exact)
  int c = i & 127, n = i >> 7;
  float v = (c < 96) ? Wseq[seq[n] * 96 + c] : Wsec[sec[n] * 32 + (c - 96)];
  hbf[i] = f2bf(v);
}

// ---------------- weight pack: f32 [128 x Ncols] -> bf16 transposed [Ncols x 128]
__global__ __launch_bounds__(256)
void pack_wT(const float* __restrict__ B, unsigned short* __restrict__ Bt, int Ncols) {
  int o = blockIdx.x * 256 + threadIdx.x;     // over Ncols*128 (exact)
  int n = o >> 7, k = o & 127;
  Bt[o] = f2bf(B[(size_t)k * Ncols + n]);
}

// ---------------- WMMA GEMM: C[M x Ncols] = A_bf16[M x 128] * Bt_bf16[Ncols x 128]^T
// One wave per 16 x (16*TC) output tile; A fragment reused across TC WMMAs.
template <int TC>
__global__ __launch_bounds__(32)
void wmma_gemm(const unsigned short* __restrict__ A, const unsigned short* __restrict__ Bt,
               const float* __restrict__ bias, float* __restrict__ C, int Ncols) {
  const int lane  = threadIdx.x;
  const int half  = lane >> 4;
  const int l15   = lane & 15;
  const int mBase = blockIdx.x * 16;
  const int nBase = blockIdx.y * (16 * TC);
  const unsigned short* arow = A + (size_t)(mBase + l15) * D;
  v8f acc[TC];
#pragma unroll
  for (int j = 0; j < TC; ++j) acc[j] = (v8f){};
#pragma unroll
  for (int kb = 0; kb < D; kb += 32) {
    // A 16x32 bf16 lane layout: two contiguous 8-elem runs per lane
    //   lanes0-15: K = kb+{0..7}, kb+{16..23}; lanes16-31: +8
    Frag a;
    a.q[0] = *(const u32x4*)(arow + kb + half * 8);
    a.q[1] = *(const u32x4*)(arow + kb + 16 + half * 8);
#pragma unroll
    for (int j = 0; j < TC; ++j) {
      // B 32x16 lane layout: lanes0-15 K=kb+{0..7},kb+{16..23}; lanes16-31 +8 (col n)
      const unsigned short* brow = Bt + (size_t)(nBase + j * 16 + l15) * D;
      Frag b;
      b.q[0] = *(const u32x4*)(brow + kb + half * 8);
      b.q[1] = *(const u32x4*)(brow + kb + 16 + half * 8);
      acc[j] = __builtin_amdgcn_wmma_f32_16x16x32_bf16(false, a.v, false, b.v,
                                                       (short)0, acc[j], false, false);
    }
  }
#pragma unroll
  for (int j = 0; j < TC; ++j) {
    int n = nBase + j * 16 + l15;
    float bi = bias ? bias[n] : 0.0f;
#pragma unroll
    for (int r = 0; r < 8; ++r) {                // C: vgpr r -> row half*8+r, col n
      int row = mBase + half * 8 + r;
      C[(size_t)row * Ncols + n] = acc[j][r] + bi;
    }
  }
}

// ---------------- per-edge feature + logits (one wave32 per edge) ----------------
__global__ __launch_bounds__(256)
void edge_kernel(const float* __restrict__ ni, const float* __restrict__ nj,
                 const float* __restrict__ e, int e_dim,
                 const float* __restrict__ Wf, const float* __restrict__ ebias,
                 const float* __restrict__ attn,
                 const int* __restrict__ src, const int* __restrict__ dst,
                 int heads, float* __restrict__ f_out, float* __restrict__ logits) {
  __shared__ float sWf[HE * HE];
  const int lane = threadIdx.x;                       // 0..31 == channel
  const int tid  = threadIdx.y * 32 + lane;
  if (e_dim == 1) {
    if (tid < HE) sWf[tid] = Wf[tid];
  } else {
    for (int idx = tid; idx < HE * HE; idx += 256) sWf[idx] = Wf[idx];
  }
  __syncthreads();

  const int ei = blockIdx.x * 8 + threadIdx.y;        // NE/8 blocks exact
  const int s = src[ei], d = dst[ei];
  float val = ni[(size_t)s * HE + lane] + nj[(size_t)d * HE + lane] + ebias[lane];
  if (e_dim == 1) {
    val += e[ei] * sWf[lane];                         // W0_fij is 1x32
  } else {
    float ev = e[(size_t)ei * HE + lane];
#pragma unroll
    for (int k = 0; k < HE; ++k)                      // e(1x32) @ Wfij(32x32)
      val += __shfl(ev, k, 32) * sWf[k * HE + lane];
  }
  val = val > 0.f ? val : 0.01f * val;                // leaky relu
  f_out[(size_t)ei * HE + lane] = val;
  float p = val * attn[lane];                         // per-head dot
  int gs = HE / heads;                                // 32 or 8
  for (int off = gs >> 1; off; off >>= 1) p += __shfl_xor(p, off, 32);
  if ((lane & (gs - 1)) == 0) logits[(size_t)ei * heads + (lane / gs)] = p;
}

// ---------------- segmented softmax pieces ----------------
__global__ __launch_bounds__(256)
void seg_max_kernel(const float* __restrict__ logits, const int* __restrict__ dst,
                    unsigned* __restrict__ segmax, int hshift, int hmask) {
  int t = blockIdx.x * 256 + threadIdx.x;             // over NE*heads (exact)
  int e = t >> hshift, h = t & hmask;
  atomicMax(&segmax[((unsigned)dst[e] << hshift) | h], enc_f32(logits[t]));
}

__global__ __launch_bounds__(256)
void seg_z_kernel(float* __restrict__ logits, const int* __restrict__ dst,
                  const unsigned* __restrict__ segmax, float* __restrict__ segsum,
                  int hshift, int hmask) {
  int t = blockIdx.x * 256 + threadIdx.x;
  int e = t >> hshift, h = t & hmask;
  int base = ((int)dst[e] << hshift) | h;
  float z = expf(logits[t] - dec_f32(segmax[base]));
  logits[t] = z;                                       // in-place: z overwrites logit
  atomicAdd(&segsum[base], z);
}

// ---------------- aggregation: h_out[dst] += hp[src] * a -------------------------
__global__ __launch_bounds__(256)
void agg_kernel(const float* __restrict__ hp, const float* __restrict__ z,
                const float* __restrict__ segsum, const int* __restrict__ src,
                const int* __restrict__ dst, int heads, float* __restrict__ hout) {
  long t = (long)blockIdx.x * 256 + threadIdx.x;       // NE*128 (exact)
  int c = (int)(t & 127);
  int e = (int)(t >> 7);
  int head = (c * heads) >> 7;                         // c / (128/heads)
  int d = dst[e], s = src[e];
  float a = z[(size_t)e * heads + head] / segsum[(size_t)d * heads + head];
  atomicAdd(&hout[(size_t)d * D + c], hp[(size_t)s * D + c] * a);
}

// ---------------- batchnorm stats (sum, sumsq per channel) -----------------------
__global__ __launch_bounds__(256)
void bn_stats(const float* __restrict__ x, long total, int cmask,
              float* __restrict__ gsum, float* __restrict__ gsumsq) {
  __shared__ float ls[128], lss[128];
  int tid = threadIdx.x;
  if (tid < 128) { ls[tid] = 0.f; lss[tid] = 0.f; }
  __syncthreads();
  float s = 0.f, ss = 0.f;
  long start = (long)blockIdx.x * 256 + tid;
  long stride = (long)gridDim.x * 256;                 // multiple of 128 -> channel fixed
  for (long i = start; i < total; i += stride) { float v = x[i]; s += v; ss += v * v; }
  int c = (int)(start & cmask);
  atomicAdd(&ls[c], s); atomicAdd(&lss[c], ss);        // LDS float atomics (ds_add_f32)
  __syncthreads();
  if (tid <= cmask) { atomicAdd(&gsum[tid], ls[tid]); atomicAdd(&gsumsq[tid], lss[tid]); }
}

// ---------------- batchnorm apply + leaky relu (+ optional bf16 mirror) ----------
__global__ __launch_bounds__(256)
void bn_apply(const float* __restrict__ xin, const float* __restrict__ sum,
              const float* __restrict__ sumsq, const float* __restrict__ gamma,
              const float* __restrict__ beta, float cntInv, int cmask,
              float* __restrict__ xout, unsigned short* __restrict__ xbf) {
  long i = (long)blockIdx.x * 256 + threadIdx.x;       // total divisible by 256
  int c = (int)(i & cmask);
  float mu  = sum[c] * cntInv;
  float var = sumsq[c] * cntInv - mu * mu;
  float v = (xin[i] - mu) * rsqrtf(var + 1e-5f) * gamma[c] + beta[c];
  v = v > 0.f ? v : 0.01f * v;
  xout[i] = v;
  if (xbf) xbf[i] = f2bf(v);
}

__global__ __launch_bounds__(256)
void cvt_bf16(const float* __restrict__ x, unsigned short* __restrict__ y) {
  long i = (long)blockIdx.x * 256 + threadIdx.x;
  y[i] = f2bf(x[i]);
}

// ---------------- final FC: out = h @ Wfc + bfc ----------------------------------
__global__ __launch_bounds__(256)
void final_fc(const float* __restrict__ h, const float* __restrict__ Wfc,
              const float* __restrict__ bfc, float* __restrict__ out) {
  int n = blockIdx.x * 256 + threadIdx.x;
  if (n >= NN) return;
  float s = bfc[0];
  const float* row = h + (size_t)n * D;
#pragma unroll 4
  for (int c = 0; c < D; ++c) s += row[c] * Wfc[c];
  out[n] = s;
}

extern "C" void kernel_launch(void* const* d_in, const int* in_sizes, int n_in,
                              void* d_out, int out_size, void* d_ws, size_t ws_size,
                              hipStream_t stream) {
  (void)in_sizes; (void)n_in; (void)out_size; (void)ws_size;
  const int*   seq     = (const int*)d_in[0];
  const int*   sec     = (const int*)d_in[1];
  const float* edge_p  = (const float*)d_in[2];
  const int*   src     = (const int*)d_in[3];
  const int*   dst     = (const int*)d_in[4];
  const float* Wseq    = (const float*)d_in[5];
  const float* Wsec    = (const float*)d_in[6];
  const float* W0_node = (const float*)d_in[7];
  const float* b0_node = (const float*)d_in[8];
  const float* W0_ni   = (const float*)d_in[9];
  const float* W0_fij  = (const float*)d_in[10];
  const float* W0_nj   = (const float*)d_in[11];
  const float* attn0   = (const float*)d_in[12];
  const float* ebias0  = (const float*)d_in[13];
  const float* Wn      = (const float*)d_in[14];
  const float* bnb     = (const float*)d_in[15];
  const float* Wni     = (const float*)d_in[16];
  const float* Wfij    = (const float*)d_in[17];
  const float* Wnj     = (const float*)d_in[18];
  const float* attnB   = (const float*)d_in[19];
  const float* ebiasB  = (const float*)d_in[20];
  const float* gn      = (const float*)d_in[21];
  const float* betan   = (const float*)d_in[22];
  const float* ge      = (const float*)d_in[23];
  const float* betae   = (const float*)d_in[24];
  const float* Wfc     = (const float*)d_in[25];
  const float* bfc     = (const float*)d_in[26];

  // ---- workspace layout ----
  size_t off = 0;
  char* base = (char*)d_ws;
  auto alloc = [&](size_t bytes) -> void* {
    void* p = base + off;
    off += (bytes + 255) & ~(size_t)255;
    return p;
  };
  float*          h_f32  = (float*)alloc((size_t)NN * D * 4);   // normalized h (f32)
  unsigned short* h_bf16 = (unsigned short*)alloc((size_t)NN * D * 2);
  float*          h_out  = (float*)alloc((size_t)NN * D * 4);   // aggregation target
  float*          hp     = (float*)alloc((size_t)NN * D * 4);
  float*          ni     = (float*)alloc((size_t)NN * HE * 4);
  float*          nj     = (float*)alloc((size_t)NN * HE * 4);
  float*          ebuf0  = (float*)alloc((size_t)NE * HE * 4);
  float*          ebuf1  = (float*)alloc((size_t)NE * HE * 4);
  float*          zlog   = (float*)alloc((size_t)NE * 4 * 4);   // logits -> z, in place
  unsigned*       segmax = (unsigned*)alloc((size_t)NN * 4 * 4);
  float*          segsum = (float*)alloc((size_t)NN * 4 * 4);
  float*          stats  = (float*)alloc(320 * 4);              // [128 sum|128 sq|32 sum|32 sq]
  // packed bf16 transposed weights
  unsigned short* w0ni_t   = (unsigned short*)alloc((size_t)HE * D * 2);
  unsigned short* w0nj_t   = (unsigned short*)alloc((size_t)HE * D * 2);
  unsigned short* w0node_t = (unsigned short*)alloc((size_t)D * D * 2);
  unsigned short* wni_t    = (unsigned short*)alloc((size_t)DEPTH * HE * D * 2);
  unsigned short* wnj_t    = (unsigned short*)alloc((size_t)DEPTH * HE * D * 2);
  unsigned short* wn_t     = (unsigned short*)alloc((size_t)DEPTH * D * D * 2);
  float* sh  = stats;       float* sqh = stats + 128;
  float* se  = stats + 256; float* sqe = stats + 288;

  const dim3 b32(32), b256(256), bEdge(32, 8);

  // ---- pack all weights to transposed bf16 ----
  pack_wT<<<HE * D / 256, b256, 0, stream>>>(W0_ni, w0ni_t, HE);
  pack_wT<<<HE * D / 256, b256, 0, stream>>>(W0_nj, w0nj_t, HE);
  pack_wT<<<D * D / 256,  b256, 0, stream>>>(W0_node, w0node_t, D);
  for (int l = 0; l < DEPTH; ++l) {
    pack_wT<<<HE * D / 256, b256, 0, stream>>>(Wni + (size_t)l * D * HE, wni_t + (size_t)l * HE * D, HE);
    pack_wT<<<HE * D / 256, b256, 0, stream>>>(Wnj + (size_t)l * D * HE, wnj_t + (size_t)l * HE * D, HE);
    pack_wT<<<D * D / 256,  b256, 0, stream>>>(Wn  + (size_t)l * D * D,  wn_t  + (size_t)l * D * D,  D);
  }

  // ---- embedding ----
  embed_kernel<<<NN * D / 256, b256, 0, stream>>>(seq, sec, Wseq, Wsec, h_bf16);

  // ---- layer 0 (heads = 1) ----
  wmma_gemm<2><<<dim3(NN / 16, 1), b32, 0, stream>>>(h_bf16, w0ni_t, nullptr, ni, HE);
  wmma_gemm<2><<<dim3(NN / 16, 1), b32, 0, stream>>>(h_bf16, w0nj_t, nullptr, nj, HE);
  wmma_gemm<4><<<dim3(NN / 16, 2), b32, 0, stream>>>(h_bf16, w0node_t, b0_node, hp, D);
  edge_kernel<<<NE / 8, bEdge, 0, stream>>>(ni, nj, edge_p, 1, W0_fij, ebias0, attn0,
                                            src, dst, 1, ebuf0, zlog);
  hipMemsetAsync(segmax, 0, (size_t)NN * 4, stream);
  hipMemsetAsync(segsum, 0, (size_t)NN * 4, stream);
  hipMemsetAsync(h_f32,  0, (size_t)NN * D * 4, stream);
  seg_max_kernel<<<NE / 256, b256, 0, stream>>>(zlog, dst, segmax, 0, 0);
  seg_z_kernel  <<<NE / 256, b256, 0, stream>>>(zlog, dst, segmax, segsum, 0, 0);
  agg_kernel<<<(long)NE * D / 256, b256, 0, stream>>>(hp, zlog, segsum, src, dst, 1, h_f32);
  cvt_bf16<<<NN * D / 256, b256, 0, stream>>>(h_f32, h_bf16);

  float* e_cur = ebuf0;   // edge features for next layer
  float* f_cur = ebuf1;   // scratch for this layer's f

  // ---- 8 blocks (heads = 4) ----
  for (int l = 0; l < DEPTH; ++l) {
    const float* Wfij_l = Wfij  + (size_t)l * HE * HE;
    const float* attn_l = attnB + (size_t)l * HE;
    const float* eb_l   = ebiasB+ (size_t)l * HE;

    wmma_gemm<2><<<dim3(NN / 16, 1), b32, 0, stream>>>(h_bf16, wni_t + (size_t)l * HE * D,
                                                       nullptr, ni, HE);
    wmma_gemm<2><<<dim3(NN / 16, 1), b32, 0, stream>>>(h_bf16, wnj_t + (size_t)l * HE * D,
                                                       nullptr, nj, HE);
    wmma_gemm<4><<<dim3(NN / 16, 2), b32, 0, stream>>>(h_bf16, wn_t + (size_t)l * D * D,
                                                       bnb + (size_t)l * D, hp, D);

    edge_kernel<<<NE / 8, bEdge, 0, stream>>>(ni, nj, e_cur, HE, Wfij_l, eb_l, attn_l,
                                              src, dst, 4, f_cur, zlog);

    hipMemsetAsync(segmax, 0, (size_t)NN * 4 * 4, stream);
    hipMemsetAsync(segsum, 0, (size_t)NN * 4 * 4, stream);
    hipMemsetAsync(h_out,  0, (size_t)NN * D * 4, stream);
    seg_max_kernel<<<NE * 4 / 256, b256, 0, stream>>>(zlog, dst, segmax, 2, 3);
    seg_z_kernel  <<<NE * 4 / 256, b256, 0, stream>>>(zlog, dst, segmax, segsum, 2, 3);
    agg_kernel<<<(long)NE * D / 256, b256, 0, stream>>>(hp, zlog, segsum, src, dst, 4, h_out);

    hipMemsetAsync(stats, 0, 320 * 4, stream);
    bn_stats<<<1024, b256, 0, stream>>>(h_out, (long)NN * D, 127, sh, sqh);
    bn_apply<<<NN * D / 256, b256, 0, stream>>>(h_out, sh, sqh, gn + l * D, betan + l * D,
                                                1.0f / NN, 127, h_f32, h_bf16);
    bn_stats<<<1024, b256, 0, stream>>>(f_cur, (long)NE * HE, 31, se, sqe);
    bn_apply<<<(long)NE * HE / 256, b256, 0, stream>>>(f_cur, se, sqe, ge + l * HE,
                                                       betae + l * HE, 1.0f / NE, 31,
                                                       f_cur, nullptr);
    float* t = e_cur; e_cur = f_cur; f_cur = t;
  }

  // ---- final FC ----
  final_fc<<<(NN + 255) / 256, b256, 0, stream>>>(h_f32, Wfc, bfc, (float*)d_out);
}
